// Critic_Mix_67456756351048
// MI455X (gfx1250) — matrix operand order repaired and checked by reading, not compiled, not run on hardware
//
#include <hip/hip_runtime.h>
#include <hip/hip_bf16.h>

typedef __attribute__((ext_vector_type(16))) _Float16 v16h;
typedef __attribute__((ext_vector_type(8)))  _Float16 v8h;
typedef __attribute__((ext_vector_type(8)))  float    v8f;

#define BATCH   262144
#define MT      32          // batch rows per workgroup (2 waves x 16 rows)
#define STRIDE  136         // LDS row stride in halves (128 + 8 pad, 16B aligned)
#define SBUF    (MT * STRIDE)   // halves per activation stream buffer
#define HOFF    204800      // per-head offset (halves) in converted-weight workspace

struct KParams {
  const float* x;
  const float* u;
  const float* mixf;
  const float* P;
  const float* bMain[2][3];   // b1,b2,b3 / b5,b6,b7
  const float* tB[2][3];      // tb1,tb2,tb3 / tb5,tb6,tb7  ([K,O] flat)
  const float* W4[2];         // W4 / W8  (f32 [1,64])
  const float* b4[2];
  const _Float16* cW[2][3];   // converted main weights [O][128]
  const _Float16* ctW[2][3];  // converted teacher weights [K][O][128]
  float* out;
};

// ---- fragment helpers (documented gfx1250 WMMA VGPR layouts) ----
// All LDS accesses are expressed as (shared base + integer offset) so the
// address-space inference keeps them on the ds_* path (no flat ops).

// 16-bit A matrix 16x32: lanes 0-15 row=lane, k halves {0..7,16..23};
// lanes 16-31 row=lane-16, k halves {8..15,24..31}.
__device__ __forceinline__ v16h load_A(const _Float16* smem, int off, int lane, int kstep) {
  int row = lane & 15;
  int k0  = kstep * 32 + ((lane & 16) ? 8 : 0);
  const _Float16* p = smem + off + row * STRIDE + k0;
  v8h lo = *(const v8h*)(p);
  v8h hi = *(const v8h*)(p + 16);
  v16h a;
#pragma unroll
  for (int i = 0; i < 8; ++i) { a[i] = lo[i]; a[i + 8] = hi[i]; }
  return a;
}

// 16-bit B matrix 32x16: lanes 0-15 col=lane, k halves {0..7,16..23};
// lanes 16-31 col=lane-16, k halves {8..15,24..31}. Weights row-major [o][i], i contiguous.
__device__ __forceinline__ v16h load_B(const _Float16* __restrict__ w, int lane, int nt, int kstep) {
  int n  = nt * 16 + (lane & 15);
  int k0 = kstep * 32 + ((lane & 16) ? 8 : 0);
  const _Float16* p = w + n * 128 + k0;
  v8h lo = *(const v8h*)(p);
  v8h hi = *(const v8h*)(p + 16);
  v16h b;
#pragma unroll
  for (int i = 0; i < 8; ++i) { b[i] = lo[i]; b[i + 8] = hi[i]; }
  return b;
}

// f32 C/D 16x16: lane l vgpr r -> (m = r + (l>=16?8:0), n = l&15)
__device__ __forceinline__ void store_D(_Float16* smem, int off, int lane, v8f d) {
  int n  = lane & 15;
  int mb = (lane & 16) ? 8 : 0;
#pragma unroll
  for (int r = 0; r < 8; ++r) smem[off + (mb + r) * STRIDE + n] = (_Float16)d[r];
}

__device__ __forceinline__ v8f wmma_step(v16h a, v16h b, v8f c) {
  return __builtin_amdgcn_wmma_f32_16x16x32_f16(false, a, false, b, (short)0, c, false, false);
}

__device__ __forceinline__ v8f v8f_zero() {
  v8f z = {0.f, 0.f, 0.f, 0.f, 0.f, 0.f, 0.f, 0.f};
  return z;
}

// One fused layer: 4 teacher GEMMs + main GEMM (K=128, out cols = NT*16),
// teacher mix (pre-ReLU) into main, ReLU everywhere, in-place LDS update.
// Teacher stream t uses LDS offsets inT0 + t*inTs / outT0 + t*outTs (affine,
// so the rolled loop keeps pure integer addressing into the shared base).
template<int NT>
__device__ __forceinline__ void do_layer(
    _Float16* smem, int lane,
    int inMainOff, int inT0, int inTs,
    int outMainOff, int outT0, int outTs,
    const _Float16* __restrict__ wMain, const _Float16* __restrict__ wT,
    const float* __restrict__ bMain, const float* __restrict__ bT,
    const float* __restrict__ P, float m, float one_m)
{
  v8f mix[NT];
#pragma unroll
  for (int i = 0; i < NT; ++i) mix[i] = v8f_zero();

#pragma unroll 1
  for (int t = 0; t < 4; ++t) {
    const int inOff  = inT0 + t * inTs;
    const int outOff = outT0 + t * outTs;
    v16h a0 = load_A(smem, inOff, lane, 0);
    v16h a1 = load_A(smem, inOff, lane, 1);
    v16h a2 = load_A(smem, inOff, lane, 2);
    v16h a3 = load_A(smem, inOff, lane, 3);
    const _Float16* w = wT + (size_t)t * (NT * 16 * 128);
    const float Pt = P[t];
#pragma unroll
    for (int nt = 0; nt < NT; ++nt) {
      v8f acc = v8f_zero();
      acc = wmma_step(a0, load_B(w, lane, nt, 0), acc);
      acc = wmma_step(a1, load_B(w, lane, nt, 1), acc);
      acc = wmma_step(a2, load_B(w, lane, nt, 2), acc);
      acc = wmma_step(a3, load_B(w, lane, nt, 3), acc);
      float bv = bT[t * (NT * 16) + nt * 16 + (lane & 15)];
      v8f hr;
#pragma unroll
      for (int r = 0; r < 8; ++r) {
        float ht = acc[r] + bv;          // pre-ReLU teacher output
        mix[nt][r] += ht * Pt;           // mixsum uses pre-ReLU values
        hr[r] = fmaxf(ht, 0.0f);         // teacher stream continues ReLU'd
      }
      store_D(smem, outOff, lane, hr);
    }
  }
  // main stream
  {
    v16h a0 = load_A(smem, inMainOff, lane, 0);
    v16h a1 = load_A(smem, inMainOff, lane, 1);
    v16h a2 = load_A(smem, inMainOff, lane, 2);
    v16h a3 = load_A(smem, inMainOff, lane, 3);
#pragma unroll
    for (int nt = 0; nt < NT; ++nt) {
      v8f acc = v8f_zero();
      acc = wmma_step(a0, load_B(wMain, lane, nt, 0), acc);
      acc = wmma_step(a1, load_B(wMain, lane, nt, 1), acc);
      acc = wmma_step(a2, load_B(wMain, lane, nt, 2), acc);
      acc = wmma_step(a3, load_B(wMain, lane, nt, 3), acc);
      float bv = bMain[nt * 16 + (lane & 15)];
      v8f h;
#pragma unroll
      for (int r = 0; r < 8; ++r)
        h[r] = fmaxf((acc[r] + bv) * one_m + m * mix[nt][r], 0.0f);
      store_D(smem, outMainOff, lane, h);
    }
  }
}

__global__ __launch_bounds__(64) void critic_fused(KParams p) {
  __shared__ __align__(16) _Float16 smem[6 * SBUF];  // 5 streams + xu, 52224 B

  const int head = (int)blockIdx.y;
  const int tid  = (int)threadIdx.x;
  const int wave = tid >> 5;
  const int lane = tid & 31;
  const int rBase = wave * 16;
  const long gRow0 = (long)blockIdx.x * MT + rBase;

  const int waveOff = rBase * STRIDE;       // wave-private row block
  const int act0    = 0 * SBUF + waveOff;   // main stream
  const int act1    = 1 * SBUF + waveOff;   // teacher streams: act1 + t*SBUF
  const int xuOff   = 5 * SBUF + waveOff;

  const float m = p.mixf[0];
  const float one_m = 1.0f - m;

  // stage xu = concat(x[:,0:96], u[:,0:32]) for this wave's 16 rows (f32 -> f16)
  for (int idx = lane; idx < 16 * 128; idx += 32) {
    int r = idx >> 7;
    int c = idx & 127;
    float v = (c < 96) ? p.x[(gRow0 + r) * 96 + c]
                       : p.u[(gRow0 + r) * 32 + (c - 96)];
    smem[xuOff + r * STRIDE + c] = (_Float16)v;
  }
  // rows are wave-private; in-wave LDS ordering suffices (no barrier needed)

  // L1: 128 -> 128, teachers read shared xu (teacher input stride 0)
  do_layer<8>(smem, lane, xuOff, xuOff, 0, act0, act1, SBUF,
              p.cW[head][0], p.ctW[head][0], p.bMain[head][0], p.tB[head][0],
              p.P, m, one_m);
  // L2: 128 -> 128, in-place
  do_layer<8>(smem, lane, act0, act1, SBUF, act0, act1, SBUF,
              p.cW[head][1], p.ctW[head][1], p.bMain[head][1], p.tB[head][1],
              p.P, m, one_m);
  // L3: 128 -> 64, in-place
  do_layer<4>(smem, lane, act0, act1, SBUF, act0, act1, SBUF,
              p.cW[head][2], p.ctW[head][2], p.bMain[head][2], p.tB[head][2],
              p.P, m, one_m);

  // L4: main stream 64 -> 1 (f32, no ReLU, no mixing)
  if (lane < 16) {
    const float* w4 = p.W4[head];
    float s = p.b4[head][0];
#pragma unroll
    for (int c = 0; c < 64; ++c)
      s += (float)smem[act0 + lane * STRIDE + c] * w4[c];
    p.out[(size_t)head * BATCH + gRow0 + lane] = s;
  }
}

__global__ void cvt_f32_f16(const float* __restrict__ s, _Float16* __restrict__ d, int n) {
  int i = (int)blockIdx.x * 256 + (int)threadIdx.x;
  if (i < n) d[i] = (_Float16)s[i];
}

extern "C" void kernel_launch(void* const* d_in, const int* in_sizes, int n_in,
                              void* d_out, int out_size, void* d_ws, size_t ws_size,
                              hipStream_t stream) {
  (void)in_sizes; (void)n_in; (void)out_size; (void)ws_size;
  _Float16* ws = (_Float16*)d_ws;

  // Convert weights f32 -> f16 into workspace.
  // Per-head layout (halves): cWa 16384 | cWb 16384 | cWc 8192 | tWa 65536 | tWb 65536 | tWc 32768
  struct Seg { int src; size_t dst; int n; };
  const Seg segs[12] = {
    {4,          0, 16384}, {6,      16384, 16384}, {8,      32768,  8192},
    {20,     40960, 65536}, {22,    106496, 65536}, {24,    172032, 32768},
    {12, HOFF +     0, 16384}, {14, HOFF + 16384, 16384}, {16, HOFF + 32768,  8192},
    {26, HOFF + 40960, 65536}, {28, HOFF +106496, 65536}, {30, HOFF +172032, 32768},
  };
  for (int i = 0; i < 12; ++i) {
    int n = segs[i].n;
    cvt_f32_f16<<<(n + 255) / 256, 256, 0, stream>>>(
        (const float*)d_in[segs[i].src], ws + segs[i].dst, n);
  }

  KParams p{};
  p.x    = (const float*)d_in[0];
  p.u    = (const float*)d_in[1];
  p.mixf = (const float*)d_in[2];
  p.P    = (const float*)d_in[3];
  p.bMain[0][0] = (const float*)d_in[5];
  p.bMain[0][1] = (const float*)d_in[7];
  p.bMain[0][2] = (const float*)d_in[9];
  p.bMain[1][0] = (const float*)d_in[13];
  p.bMain[1][1] = (const float*)d_in[15];
  p.bMain[1][2] = (const float*)d_in[17];
  p.W4[0] = (const float*)d_in[10];  p.b4[0] = (const float*)d_in[11];
  p.W4[1] = (const float*)d_in[18];  p.b4[1] = (const float*)d_in[19];
  p.tB[0][0] = (const float*)d_in[21];
  p.tB[0][1] = (const float*)d_in[23];
  p.tB[0][2] = (const float*)d_in[25];
  p.tB[1][0] = (const float*)d_in[27];
  p.tB[1][1] = (const float*)d_in[29];
  p.tB[1][2] = (const float*)d_in[31];
  p.cW[0][0] = ws;                 p.cW[0][1] = ws + 16384;        p.cW[0][2] = ws + 32768;
  p.ctW[0][0] = ws + 40960;        p.ctW[0][1] = ws + 106496;      p.ctW[0][2] = ws + 172032;
  p.cW[1][0] = ws + HOFF;          p.cW[1][1] = ws + HOFF + 16384; p.cW[1][2] = ws + HOFF + 32768;
  p.ctW[1][0] = ws + HOFF + 40960; p.ctW[1][1] = ws + HOFF + 106496;
  p.ctW[1][2] = ws + HOFF + 172032;
  p.out = (float*)d_out;

  dim3 grid(BATCH / MT, 2);
  critic_fused<<<grid, 64, 0, stream>>>(p);
}
